// HOGRUCell_44616120271296
// MI455X (gfx1250) — compile-verified, matrix operand order checked by reading, not run on hardware
//
#include <hip/hip_runtime.h>
#include <cstdint>
#include <cstddef>

// ---------------------------------------------------------------------------
// HOGRU cell (RK4) for MI455X / gfx1250.
//  - GEMMs: v_wmma_f32_16x16x32_bf16 (wave32, f32 accum).
//  - Global->LDS staging: GLOBAL_LOAD_ASYNC_TO_LDS_B128 (ASYNCcnt) with
//    double-buffered tiles: prefetch tile i+1 while computing tile i.
//  - lx = x0 @ W_lx^T hoisted out of the RK stages (h-independent).
//  - Each RK stage = one fused kernel: 3 GEMMs share one h tile + full gate
//    epilogue (sigmoid/sigmoid/tanh, RK accumulation) in registers.
// ---------------------------------------------------------------------------

#define B_ROWS  16384
#define K_IN    256
#define K_STATE 1024
#define K_OUT   256

typedef __bf16 bf16;
typedef __attribute__((ext_vector_type(16))) __bf16 v16bf;
typedef __attribute__((ext_vector_type(8)))  float  v8f;

#define LDS_STRIDE 40   // 32 + 8 bf16 pad: 80B rows -> 16B aligned, conflict-free

__device__ __forceinline__ bf16 f2bf(float f) {
  union { float f; unsigned u; } a; a.f = f;
  unsigned u = a.u + (0x7FFFu + ((a.u >> 16) & 1u));   // round-to-nearest-even
  unsigned short s = (unsigned short)(u >> 16);
  bf16 r; __builtin_memcpy(&r, &s, sizeof(r));
  return r;
}

__device__ __forceinline__ float sigmoid_f(float x) {
  return 1.0f / (1.0f + __expf(-x));
}

__device__ __forceinline__ v8f zero_v8f() {
  v8f z;
  #pragma unroll
  for (int i = 0; i < 8; ++i) z[i] = 0.0f;
  return z;
}

// Truncating a generic (flat) LDS pointer gives the 32-bit LDS byte address
// (ISA 10.2).
__device__ __forceinline__ unsigned lds_addr(const void* p) {
  return (unsigned)(uintptr_t)p;
}

// Async DMA of one 16B chunk: LDS[vdst-lane-addr] = MEM[vaddr]. ASYNCcnt.
__device__ __forceinline__ void async_load_b128(unsigned lds_byte_addr,
                                                const void* gptr) {
  asm volatile("global_load_async_to_lds_b128 %0, %1, off"
               :: "v"(lds_byte_addr), "v"(gptr)
               : "memory");
}

template <int N>
__device__ __forceinline__ void wait_async_le() {
#if __has_builtin(__builtin_amdgcn_s_wait_asynccnt)
  __builtin_amdgcn_s_wait_asynccnt(N);
#else
  asm volatile("s_wait_asynccnt %0" :: "i"(N) : "memory");
#endif
}

// 16x32 bf16 WMMA fragment from an LDS tile of row-major rows.
// A: row -> M, elements -> K.  B: row -> N (row of W[N,K]), elements -> K.
// Per CDNA5 ISA: lane(l<16) holds K {half*8..+7, 16+half*8..+7} of row (l&15).
__device__ __forceinline__ v16bf load_frag(const bf16* __restrict__ base,
                                           int row, int stride) {
  const int lane = threadIdx.x & 31;
  const int half = lane >> 4;
  const bf16* p = base + (size_t)(row + (lane & 15)) * stride + half * 8;
  union { v16bf v; uint4 q[2]; } u;
  u.q[0] = *(const uint4*)(p);        // K = half*8 .. half*8+7
  u.q[1] = *(const uint4*)(p + 16);   // K = 16+half*8 .. +7
  return u.v;
}

// ---------------------------------------------------------------------------
// Generic bf16 GEMM:  out[M,N] = epilogue(A[M,K] @ W[N,K]^T)
// Block 256 thr = 8 waves (4x2), block tile 128x128, wave tile 32x64, BK=32.
// Tile = 1024 x 16B chunks (A:512, B:512) -> 4 async chunks per thread.
// ---------------------------------------------------------------------------
enum { EPI_F32 = 0, EPI_TANH_BF16 = 1, EPI_BIAS_TANH_BF16 = 2 };

#define GB_ROWB   (LDS_STRIDE * 2)          // 80B per LDS row
#define GB_ABYTES (128 * GB_ROWB)           // 10240
#define GB_BUFBYTES (2 * GB_ABYTES)         // 20480 per buffer (A+B)
#define GB_BUFELEMS (GB_BUFBYTES / 2)

__device__ __forceinline__ void gemm_issue_tile(
    const bf16* __restrict__ A, const bf16* __restrict__ W,
    int K, int mBase, int nBase, int k0, unsigned ldsBase, int tid) {
  #pragma unroll
  for (int r = 0; r < 4; ++r) {
    const int c = tid + r * 256;              // 0..1023, branch-uniform per r
    if (c < 512) {                            // A: row = c/4, 4 chunks/row
      const int row = c >> 2, sg = c & 3;
      async_load_b128(ldsBase + row * GB_ROWB + sg * 16,
                      A + (size_t)(mBase + row) * K + k0 + sg * 8);
    } else {                                  // B
      const int cb = c - 512;
      const int row = cb >> 2, sg = cb & 3;
      async_load_b128(ldsBase + GB_ABYTES + row * GB_ROWB + sg * 16,
                      W + (size_t)(nBase + row) * K + k0 + sg * 8);
    }
  }
}

template <int EPI>
__global__ __launch_bounds__(256)
void gemm_bf16_kernel(const bf16* __restrict__ A,
                      const bf16* __restrict__ W,
                      const float* __restrict__ bias,
                      void* __restrict__ out,
                      int M, int N, int K) {
  __shared__ __attribute__((aligned(16))) bf16 smem[2 * GB_BUFELEMS];

  const int mBase = blockIdx.x * 128;
  const int nBase = blockIdx.y * 128;
  const int tid   = threadIdx.x;
  const int wave  = tid >> 5;
  const int wm    = (wave >> 1) * 32;
  const int wn    = (wave &  1) * 64;
  const unsigned smemA = lds_addr(smem);

  v8f acc[2][4];
  #pragma unroll
  for (int i = 0; i < 2; ++i)
    #pragma unroll
    for (int j = 0; j < 4; ++j) acc[i][j] = zero_v8f();

  const int NT = K >> 5;
  gemm_issue_tile(A, W, K, mBase, nBase, 0, smemA, tid);

  for (int it = 0; it < NT; ++it) {
    if (it + 1 < NT) {
      gemm_issue_tile(A, W, K, mBase, nBase, (it + 1) << 5,
                      smemA + ((it + 1) & 1) * GB_BUFBYTES, tid);
      wait_async_le<4>();       // in-order: tile `it`'s 4 chunks are done
    } else {
      wait_async_le<0>();
    }
    __syncthreads();

    const bf16* bufA = smem + (it & 1) * GB_BUFELEMS;
    const bf16* bufB = bufA + 128 * LDS_STRIDE;
    v16bf a[2], b[4];
    #pragma unroll
    for (int mi = 0; mi < 2; ++mi) a[mi] = load_frag(bufA, wm + mi * 16, LDS_STRIDE);
    #pragma unroll
    for (int ni = 0; ni < 4; ++ni) b[ni] = load_frag(bufB, wn + ni * 16, LDS_STRIDE);
    #pragma unroll
    for (int mi = 0; mi < 2; ++mi)
      #pragma unroll
      for (int ni = 0; ni < 4; ++ni)
        acc[mi][ni] = __builtin_amdgcn_wmma_f32_16x16x32_bf16(
            false, a[mi], false, b[ni], (short)0, acc[mi][ni], false, false);
    __syncthreads();            // all waves done reading before buf reuse
  }

  // Epilogue. C layout: element r -> M = r + 8*(lane>=16), N = lane&15.
  const int lane = tid & 31;
  const int cRow = (lane >> 4) * 8;
  const int cCol = lane & 15;
  #pragma unroll
  for (int mi = 0; mi < 2; ++mi) {
    #pragma unroll
    for (int ni = 0; ni < 4; ++ni) {
      const int gn = nBase + wn + ni * 16 + cCol;
      const float bv = (EPI == EPI_BIAS_TANH_BF16) ? bias[gn] : 0.0f;
      #pragma unroll
      for (int r = 0; r < 8; ++r) {
        const int gm = mBase + wm + mi * 16 + cRow + r;
        const size_t idx = (size_t)gm * N + gn;
        const float v = acc[mi][ni][r] + bv;
        if (EPI == EPI_F32) ((float*)out)[idx] = v;
        else                ((bf16*)out)[idx]  = f2bf(tanhf(v));
      }
    }
  }
}

// ---------------------------------------------------------------------------
// Fused RK stage, 128x64 tile, 3 GEMMs sharing the h tile:
//   z  = sigmoid(lx[:,n]      + h@W_hg[n]^T       + b_hg[n])
//   r  = sigmoid(lx[:,K+n]    + h@W_hg[K+n]^T     + b_hg[K+n])
//   lin=         lx[:,2K+n]   + r*(h@W_hl[n]^T)   + b_hl[n]
//   k  = z*(tanh(lin) - h)                       (MEANDT == 1)
// acc += wAcc*k ; h_next = state + dt*stepMul*k (f32+bf16)
// Last stage: state_new = state + dt*(acc+k)/6 -> d_out + bf16 buffer.
// Tile = 1280 chunks (A:512, B:768) -> 5 async chunks per thread.
// ---------------------------------------------------------------------------
#define SB_ABYTES   (128 * GB_ROWB)                 // 10240
#define SB_BUFBYTES ((128 + 192) * GB_ROWB)         // 25600 per buffer
#define SB_BUFELEMS (SB_BUFBYTES / 2)

__device__ __forceinline__ void stage_issue_tile(
    const bf16* __restrict__ h_bf, const bf16* __restrict__ Whg,
    const bf16* __restrict__ Whl, int mBase, int nBase, int k0,
    unsigned ldsBase, int tid) {
  #pragma unroll
  for (int r = 0; r < 5; ++r) {
    const int c = tid + r * 256;              // 0..1279, branch-uniform per r
    if (c < 512) {                            // A: h tile, 4 chunks/row
      const int row = c >> 2, sg = c & 3;
      async_load_b128(ldsBase + row * GB_ROWB + sg * 16,
                      h_bf + (size_t)(mBase + row) * K_STATE + k0 + sg * 8);
    } else {                                  // B: 192 gathered weight rows
      const int cb = c - 512;
      const int row = cb >> 2, sg = cb & 3;   // row uniform-set per r
      const bf16* src;
      if (row < 64)       src = Whg + (size_t)(nBase + row) * K_STATE;
      else if (row < 128) src = Whg + (size_t)(K_STATE + nBase + row - 64) * K_STATE;
      else                src = Whl + (size_t)(nBase + row - 128) * K_STATE;
      async_load_b128(ldsBase + SB_ABYTES + row * GB_ROWB + sg * 16,
                      src + k0 + sg * 8);
    }
  }
}

__global__ __launch_bounds__(256)
void rk_stage_kernel(const bf16*  __restrict__ h_bf,    // [B,K]  bf16 A-matrix
                     const float* __restrict__ h_f32,   // [B,K]  (== state, stage 1)
                     const float* __restrict__ state,   // [B,K]
                     const bf16*  __restrict__ Whg,     // [2K,K] bf16
                     const bf16*  __restrict__ Whl,     // [K,K]  bf16
                     const float* __restrict__ b_hg,    // [2K]
                     const float* __restrict__ b_hl,    // [K]
                     const float* __restrict__ lx,      // [B,3K]
                     const float* __restrict__ dtp,     // [1]
                     float* __restrict__ acc_buf,       // [B,K]
                     float* __restrict__ hn_f32,        // next h / state_new out
                     bf16*  __restrict__ hn_bf,         // next h bf16
                     float stepMul, float wAcc, int isFirst, int isLast) {
  __shared__ __attribute__((aligned(16))) bf16 smem[2 * SB_BUFELEMS];

  const int mBase = blockIdx.x * 128;
  const int nBase = blockIdx.y * 64;
  const int tid   = threadIdx.x;
  const int wave  = tid >> 5;
  const int wm    = (wave >> 1) * 32;   // 4 waves in M
  const int wn    = (wave &  1) * 32;   // 2 waves in N
  const unsigned smemA = lds_addr(smem);

  v8f acc[3][2][2];
  #pragma unroll
  for (int s = 0; s < 3; ++s)
    #pragma unroll
    for (int i = 0; i < 2; ++i)
      #pragma unroll
      for (int j = 0; j < 2; ++j) acc[s][i][j] = zero_v8f();

  const int NT = K_STATE >> 5;          // 32
  stage_issue_tile(h_bf, Whg, Whl, mBase, nBase, 0, smemA, tid);

  for (int it = 0; it < NT; ++it) {
    if (it + 1 < NT) {
      stage_issue_tile(h_bf, Whg, Whl, mBase, nBase, (it + 1) << 5,
                       smemA + ((it + 1) & 1) * SB_BUFBYTES, tid);
      wait_async_le<5>();
    } else {
      wait_async_le<0>();
    }
    __syncthreads();

    const bf16* bufA = smem + (it & 1) * SB_BUFELEMS;
    const bf16* bufB = bufA + 128 * LDS_STRIDE;
    v16bf a[2];
    #pragma unroll
    for (int mi = 0; mi < 2; ++mi) a[mi] = load_frag(bufA, wm + mi * 16, LDS_STRIDE);
    #pragma unroll
    for (int s = 0; s < 3; ++s) {
      #pragma unroll
      for (int ni = 0; ni < 2; ++ni) {
        const v16bf b = load_frag(bufB + s * 64 * LDS_STRIDE, wn + ni * 16, LDS_STRIDE);
        #pragma unroll
        for (int mi = 0; mi < 2; ++mi)
          acc[s][mi][ni] = __builtin_amdgcn_wmma_f32_16x16x32_bf16(
              false, a[mi], false, b, (short)0, acc[s][mi][ni], false, false);
      }
    }
    __syncthreads();
  }

  const float dtv  = dtp[0];
  const int   lane = tid & 31;
  const int   cRow = (lane >> 4) * 8;
  const int   cCol = lane & 15;
  #pragma unroll
  for (int mi = 0; mi < 2; ++mi) {
    #pragma unroll
    for (int ni = 0; ni < 2; ++ni) {
      const int gn = nBase + wn + ni * 16 + cCol;
      const float bz = b_hg[gn];
      const float br = b_hg[K_STATE + gn];
      const float bl = b_hl[gn];
      #pragma unroll
      for (int r = 0; r < 8; ++r) {
        const int gm = mBase + wm + mi * 16 + cRow + r;
        const size_t i3 = (size_t)gm * (3 * K_STATE) + gn;
        const size_t i1 = (size_t)gm * K_STATE + gn;
        const float z   = sigmoid_f(lx[i3]               + acc[0][mi][ni][r] + bz);
        const float rr  = sigmoid_f(lx[i3 + K_STATE]     + acc[1][mi][ni][r] + br);
        const float lin = lx[i3 + 2 * K_STATE] + rr * acc[2][mi][ni][r] + bl;
        const float kv  = z * (tanhf(lin) - h_f32[i1]);
        if (isLast) {
          const float sn = state[i1] + dtv * (acc_buf[i1] + kv) * (1.0f / 6.0f);
          hn_f32[i1] = sn;
          hn_bf[i1]  = f2bf(sn);
        } else {
          const float a0 = isFirst ? 0.0f : acc_buf[i1];
          acc_buf[i1] = a0 + wAcc * kv;
          const float hn = state[i1] + (dtv * stepMul) * kv;
          hn_f32[i1] = hn;
          hn_bf[i1]  = f2bf(hn);
        }
      }
    }
  }
}

__global__ void f32_to_bf16_kernel(const float* __restrict__ in,
                                   bf16* __restrict__ out, size_t n) {
  size_t i = (size_t)blockIdx.x * blockDim.x + threadIdx.x;
  const size_t stride = (size_t)gridDim.x * blockDim.x;
  for (; i < n; i += stride) out[i] = f2bf(in[i]);
}

// ---------------------------------------------------------------------------
extern "C" void kernel_launch(void* const* d_in, const int* in_sizes, int n_in,
                              void* d_out, int out_size, void* d_ws, size_t ws_size,
                              hipStream_t stream) {
  (void)in_sizes; (void)n_in; (void)out_size; (void)ws_size;

  const float* x     = (const float*)d_in[0];
  const float* state = (const float*)d_in[1];
  const float* dtp   = (const float*)d_in[2];
  const float* W_in  = (const float*)d_in[3];
  const float* b_in  = (const float*)d_in[4];
  const float* W_lx  = (const float*)d_in[5];
  const float* W_hg  = (const float*)d_in[6];
  const float* b_hg  = (const float*)d_in[7];
  const float* W_hl  = (const float*)d_in[8];
  const float* b_hl  = (const float*)d_in[9];
  const float* W_y1  = (const float*)d_in[10];
  const float* W_y2  = (const float*)d_in[11];

  float* y_out = (float*)d_out;                              // [B, K_OUT]
  float* s_out = (float*)d_out + (size_t)B_ROWS * K_OUT;     // [B, K_STATE]

  char* p = (char*)d_ws;
  auto carve = [&](size_t bytes) -> char* {
    char* r = p; p += (bytes + 255) & ~(size_t)255; return r;
  };
  bf16* Win_bf = (bf16*)carve(2ull * K_STATE * K_IN);
  bf16* Wlx_bf = (bf16*)carve(2ull * 3 * K_STATE * K_STATE);
  bf16* Whg_bf = (bf16*)carve(2ull * 2 * K_STATE * K_STATE);
  bf16* Whl_bf = (bf16*)carve(2ull * K_STATE * K_STATE);
  bf16* Wy1_bf = (bf16*)carve(2ull * K_STATE * K_STATE);
  bf16* Wy2_bf = (bf16*)carve(2ull * K_OUT * K_STATE);
  bf16* x_bf   = (bf16*)carve(2ull * B_ROWS * K_IN);
  bf16* x0_bf  = (bf16*)carve(2ull * B_ROWS * K_STATE);  // reused for tanh(y1)
  bf16* hA_bf  = (bf16*)carve(2ull * B_ROWS * K_STATE);
  bf16* hB_bf  = (bf16*)carve(2ull * B_ROWS * K_STATE);
  float* lx    = (float*)carve(4ull * B_ROWS * 3 * K_STATE);
  float* hfA   = (float*)carve(4ull * B_ROWS * K_STATE);
  float* hfB   = (float*)carve(4ull * B_ROWS * K_STATE);
  float* accb  = (float*)carve(4ull * B_ROWS * K_STATE);

  auto cvt = [&](const float* src, bf16* dst, size_t n) {
    int grid = (int)((n + 1023) / 1024); if (grid > 4096) grid = 4096;
    f32_to_bf16_kernel<<<dim3(grid), dim3(256), 0, stream>>>(src, dst, n);
  };
  cvt(W_in, Win_bf, (size_t)K_STATE * K_IN);
  cvt(W_lx, Wlx_bf, (size_t)3 * K_STATE * K_STATE);
  cvt(W_hg, Whg_bf, (size_t)2 * K_STATE * K_STATE);
  cvt(W_hl, Whl_bf, (size_t)K_STATE * K_STATE);
  cvt(W_y1, Wy1_bf, (size_t)K_STATE * K_STATE);
  cvt(W_y2, Wy2_bf, (size_t)K_OUT * K_STATE);
  cvt(x,     x_bf,  (size_t)B_ROWS * K_IN);
  cvt(state, hA_bf, (size_t)B_ROWS * K_STATE);

  // x0 = tanh(x @ W_in^T + b_in)            [B,1024] bf16
  gemm_bf16_kernel<EPI_BIAS_TANH_BF16>
      <<<dim3(B_ROWS / 128, K_STATE / 128), 256, 0, stream>>>(
          x_bf, Win_bf, b_in, x0_bf, B_ROWS, K_STATE, K_IN);
  // lx = x0 @ W_lx^T  (hoisted: h-independent)   [B,3072] f32
  gemm_bf16_kernel<EPI_F32>
      <<<dim3(B_ROWS / 128, (3 * K_STATE) / 128), 256, 0, stream>>>(
          x0_bf, Wlx_bf, nullptr, lx, B_ROWS, 3 * K_STATE, K_STATE);

  const dim3 sgrid(B_ROWS / 128, K_STATE / 64);
  // k1: h = state;           acc  = k1;        h2 = state + 0.5*dt*k1
  rk_stage_kernel<<<sgrid, 256, 0, stream>>>(hA_bf, state, state, Whg_bf, Whl_bf,
      b_hg, b_hl, lx, dtp, accb, hfA, hB_bf, 0.5f, 1.0f, 1, 0);
  // k2:                      acc += 2*k2;      h3 = state + 0.5*dt*k2
  rk_stage_kernel<<<sgrid, 256, 0, stream>>>(hB_bf, hfA, state, Whg_bf, Whl_bf,
      b_hg, b_hl, lx, dtp, accb, hfB, hA_bf, 0.5f, 2.0f, 0, 0);
  // k3:                      acc += 2*k3;      h4 = state + dt*k3
  rk_stage_kernel<<<sgrid, 256, 0, stream>>>(hA_bf, hfB, state, Whg_bf, Whl_bf,
      b_hg, b_hl, lx, dtp, accb, hfA, hB_bf, 1.0f, 2.0f, 0, 0);
  // k4: state_new = state + dt*(acc + k4)/6  -> d_out(state) + hA_bf (bf16)
  rk_stage_kernel<<<sgrid, 256, 0, stream>>>(hB_bf, hfA, state, Whg_bf, Whl_bf,
      b_hg, b_hl, lx, dtp, accb, s_out, hA_bf, 0.0f, 0.0f, 0, 1);

  // t = tanh(state_new @ W_y1^T)   [B,1024] bf16 (reuse x0_bf)
  gemm_bf16_kernel<EPI_TANH_BF16>
      <<<dim3(B_ROWS / 128, K_STATE / 128), 256, 0, stream>>>(
          hA_bf, Wy1_bf, nullptr, x0_bf, B_ROWS, K_STATE, K_STATE);
  // y = t @ W_y2^T                  [B,256] f32 -> d_out
  gemm_bf16_kernel<EPI_F32>
      <<<dim3(B_ROWS / 128, K_OUT / 128), 256, 0, stream>>>(
          x0_bf, Wy2_bf, nullptr, y_out, B_ROWS, K_OUT, K_STATE);
}
